// SRAttentionBlock_16776142258777
// MI455X (gfx1250) — matrix operand order repaired and checked by reading, not compiled, not run on hardware
//
#include <hip/hip_runtime.h>
#include <hip/hip_bf16.h>
#include <math.h>

#define BB     4
#define CC     64
#define MVX    32768          // 32*32*32 voxels
#define NATOM  64
#define EMBD   128
#define NHEAD  8
#define SCL    0.35355339059327373f   // 1/sqrt(8)

typedef __attribute__((ext_vector_type(16))) _Float16 v16h;
typedef __attribute__((ext_vector_type(8)))  _Float16 v8h;
typedef __attribute__((ext_vector_type(8)))  float    v8f;

static __device__ __forceinline__ v16h cat8(v8h a, v8h b) {
  return __builtin_shufflevector(a, b, 0,1,2,3,4,5,6,7,8,9,10,11,12,13,14,15);
}
static __device__ __forceinline__ v8h zero8h() {
  v8h z;
#pragma unroll
  for (int i = 0; i < 8; ++i) z[i] = (_Float16)0.0f;
  return z;
}
static __device__ __forceinline__ v8f zero8f() {
  v8f z;
#pragma unroll
  for (int i = 0; i < 8; ++i) z[i] = 0.0f;
  return z;
}
static __device__ __forceinline__ v8f wmma16(v16h a, v16h b, v8f c) {
  // D = A(16x32 f16) * B(32x16 f16) + C(16x16 f32)
  return __builtin_amdgcn_wmma_f32_16x16x32_f16(false, a, false, b, (short)0, c,
                                                false, false);
}
// wave-private LDS round-trip fence: DS ops are in-order per wave on CDNA5,
// we only need the dependency counter drained before cross-lane re-reads.
static __device__ __forceinline__ void lds_fence() {
  asm volatile("s_wait_dscnt 0x0" ::: "memory");
}

// ---------------------------------------------------------------------------
// 1) x [B,C,32768] f32  ->  xt [B,32768,C] f16 (voxel-major rows, 128B each)
// ---------------------------------------------------------------------------
__global__ __launch_bounds__(256) void prep_x(const float* __restrict__ x,
                                              _Float16* __restrict__ xt) {
  int tid = blockIdx.x * 256 + threadIdx.x;
  int b = tid >> 15;
  int v = tid & (MVX - 1);
  const float* xp = x + (size_t)b * CC * MVX + v;
  _Float16* op = xt + ((size_t)(b * MVX + v)) * CC;
#pragma unroll
  for (int g = 0; g < 4; ++g) {
    v16h r;
#pragma unroll
    for (int j = 0; j < 16; ++j) r[j] = (_Float16)xp[(size_t)(g * 16 + j) * MVX];
    *(v16h*)(op + g * 16) = r;
  }
}

// ---------------------------------------------------------------------------
// 2) pack Wq / Wo / Wc into per-lane WMMA B-fragment order (f16).
//    B element (K,n): K = chunk*32 + p + 16*(lane>>4), n = lane&15.
// ---------------------------------------------------------------------------
__global__ __launch_bounds__(256) void prep_w(const float* __restrict__ Wq,
                                              const float* __restrict__ Wo,
                                              const float* __restrict__ Wc,
                                              _Float16* __restrict__ WqB,
                                              _Float16* __restrict__ WoB,
                                              _Float16* __restrict__ WcB) {
  int idx = blockIdx.x * 256 + threadIdx.x;
  if (idx < 512) {
    int m = idx >> 8;             // 0 = Wq, 1 = Wo
    int e = idx & 255;
    int chunk = e >> 7, nt = (e >> 5) & 3, lane = e & 31;
    int n = lane & 15, hi = lane >> 4;
    const float* W = m ? Wo : Wq;
    _Float16* D = (m ? WoB : WqB) + e * 16;
#pragma unroll
    for (int p = 0; p < 16; ++p) {
      int K = chunk * 32 + p + 16 * hi;                  // input channel
      D[p] = (_Float16)W[(nt * 16 + n) * CC + K];        // B[K][n] = W[n'][K]
    }
  } else if (idx < 512 + 6912) {
    int e = idx - 512;                                   // 27 taps * 256
    int tap = e >> 8;
    int r = e & 255;
    int chunk = r >> 7, nt = (r >> 5) & 3, lane = r & 31;
    int n = lane & 15, hi = lane >> 4;
    _Float16* D = WcB + e * 16;
#pragma unroll
    for (int p = 0; p < 16; ++p) {
      int ci = chunk * 32 + p + 16 * hi;
      int co = nt * 16 + n;
      D[p] = (_Float16)Wc[((size_t)(co * CC + ci)) * 27 + tap];
    }
  }
}

// ---------------------------------------------------------------------------
// 3) K/V projections (tiny).  kf row-major f16; v packed in B-fragment order:
//    vB[b][h][c2][lane][p] = v[atom = c2*32 + p + 16*(lane>>4)][h*8 + (lane&7)]
// ---------------------------------------------------------------------------
__global__ __launch_bounds__(256) void kv_kernel(const float* __restrict__ mol,
                                                 const float* __restrict__ Wk,
                                                 const float* __restrict__ bk,
                                                 const float* __restrict__ Wv,
                                                 const float* __restrict__ bv,
                                                 _Float16* __restrict__ kf,
                                                 _Float16* __restrict__ vB) {
  int t = threadIdx.x;                 // single 256-thread block: (b, atom)
  int b = t >> 6, atom = t & 63;
  const float* e = mol + ((size_t)(b * NATOM + atom)) * EMBD;
  int c2 = atom >> 5, kk = atom & 31, p = kk & 15, hi = kk >> 4;
  for (int c = 0; c < CC; ++c) {
    float ka = bk[c], va = bv[c];
    for (int j = 0; j < EMBD; ++j) {
      float ev = e[j];
      ka += ev * Wk[c * EMBD + j];
      va += ev * Wv[c * EMBD + j];
    }
    kf[((size_t)(b * NATOM + atom)) * CC + c] = (_Float16)ka;
    int h = c >> 3, d = c & 7;
    _Float16 hv = (_Float16)va;
    size_t base = (size_t)b * 8192 + (size_t)((h * 2 + c2) * 32) * 16;
    vB[base + (hi * 16 + d) * 16 + p] = hv;        // lanes n=d and n=d+8 (pad)
    vB[base + (hi * 16 + d + 8) * 16 + p] = hv;
  }
}

// ---------------------------------------------------------------------------
// 4) fused attention: Q-proj -> per-head scores/softmax/att*V -> O-proj+ReLU.
//    4 waves / 128-thread block; wave handles a 16-voxel M-tile.
// ---------------------------------------------------------------------------
__global__ __launch_bounds__(128) void attn_kernel(
    const _Float16* __restrict__ xt, const _Float16* __restrict__ kf,
    const _Float16* __restrict__ vB, const _Float16* __restrict__ WqB,
    const _Float16* __restrict__ WoB, const float* __restrict__ bq,
    const float* __restrict__ bo, const int* __restrict__ bn,
    _Float16* __restrict__ ovm) {
  __shared__ __align__(32) _Float16 k_s[NATOM * CC];   //  8 KB, per block
  __shared__ __align__(32) _Float16 v_s[8192];          // 16 KB, per block
  __shared__ __align__(32) _Float16 q_s[4][16 * CC];    //  8 KB, per wave
  __shared__ __align__(32) _Float16 a_s[4][16 * CC];
  __shared__ __align__(32) _Float16 o_s[4][16 * CC];

  int b = blockIdx.y;
  int tid = threadIdx.x;
  {
    const _Float16* kg = kf + (size_t)b * NATOM * CC;
#pragma unroll
    for (int i = 0; i < 2; ++i)
      *(v16h*)&k_s[(tid + i * 128) * 16] = *(const v16h*)&kg[(tid + i * 128) * 16];
    const _Float16* vg = vB + (size_t)b * 8192;
#pragma unroll
    for (int i = 0; i < 4; ++i)
      *(v16h*)&v_s[(tid + i * 128) * 16] = *(const v16h*)&vg[(tid + i * 128) * 16];
  }
  __syncthreads();

  int w = tid >> 5, lane = tid & 31;
  int an = lane & 15, hi = lane >> 4;
  int v0 = (blockIdx.x * 4 + w) * 16;
  int natoms = bn[b];

  float bqv[4], bov[4];
#pragma unroll
  for (int nt = 0; nt < 4; ++nt) { bqv[nt] = bq[nt * 16 + an]; bov[nt] = bo[nt * 16 + an]; }

  // ---- Q = xt @ Wq^T ----
  v8f acc[4];
#pragma unroll
  for (int nt = 0; nt < 4; ++nt) acc[nt] = zero8f();
  const _Float16* arow = xt + ((size_t)(b * MVX + v0 + an)) * CC;
#pragma unroll
  for (int chunk = 0; chunk < 2; ++chunk) {
    const _Float16* ap = arow + chunk * 32 + hi * 16;
    v16h A = *(const v16h*)ap;
#pragma unroll
    for (int nt = 0; nt < 4; ++nt) {
      v16h Bm = *(const v16h*)&WqB[((chunk * 4 + nt) * 32 + lane) * 16];
      acc[nt] = wmma16(A, Bm, acc[nt]);
    }
  }
#pragma unroll
  for (int nt = 0; nt < 4; ++nt)
#pragma unroll
    for (int e2 = 0; e2 < 8; ++e2)
      q_s[w][(e2 + 8 * hi) * CC + nt * 16 + an] = (_Float16)(acc[nt][e2] + bqv[nt]);
  lds_fence();

  // ---- per-head scores + softmax + att*V ----
#pragma unroll 1
  for (int h = 0; h < NHEAD; ++h) {
    v8h z8 = zero8h();
    v8h al = (lane < 16) ? *(const v8h*)&q_s[w][an * CC + h * 8] : z8;  // K 0..7
    v16h A = cat8(al, z8);                                              // K 8..31 pad
    v8f sc[4];
#pragma unroll
    for (int t = 0; t < 4; ++t) {
      v8h bl = (lane < 16) ? *(const v8h*)&k_s[(t * 16 + an) * CC + h * 8] : z8;
      sc[t] = wmma16(A, cat8(bl, z8), zero8f());
    }
#pragma unroll
    for (int t = 0; t < 4; ++t) {
      bool bad = (t * 16 + an) >= natoms;     // ragged atom mask
#pragma unroll
      for (int e2 = 0; e2 < 8; ++e2) {
        float vv = sc[t][e2] * SCL;
        sc[t][e2] = bad ? -3.0e38f : vv;
      }
    }
#pragma unroll
    for (int e2 = 0; e2 < 8; ++e2) {          // rows live in 16-lane groups
      float m = fmaxf(fmaxf(sc[0][e2], sc[1][e2]), fmaxf(sc[2][e2], sc[3][e2]));
#pragma unroll
      for (int s = 1; s < 16; s <<= 1) m = fmaxf(m, __shfl_xor(m, s));
      float sum = 0.f;
#pragma unroll
      for (int t = 0; t < 4; ++t) { float p = __expf(sc[t][e2] - m); sc[t][e2] = p; sum += p; }
#pragma unroll
      for (int s = 1; s < 16; s <<= 1) sum += __shfl_xor(sum, s);
      float inv = 1.0f / sum;
#pragma unroll
      for (int t = 0; t < 4; ++t) sc[t][e2] *= inv;
    }
#pragma unroll
    for (int t = 0; t < 4; ++t)
#pragma unroll
      for (int e2 = 0; e2 < 8; ++e2)
        a_s[w][(e2 + 8 * hi) * CC + t * 16 + an] = (_Float16)sc[t][e2];
    lds_fence();

    v8f ov = zero8f();
#pragma unroll
    for (int c2 = 0; c2 < 2; ++c2) {
      const _Float16* ap = &a_s[w][an * CC + c2 * 32 + hi * 16];
      v16h Aa = *(const v16h*)ap;
      v16h Bv = *(const v16h*)&v_s[((h * 2 + c2) * 32 + lane) * 16];
      ov = wmma16(Aa, Bv, ov);
    }
    if (an < 8) {
#pragma unroll
      for (int e2 = 0; e2 < 8; ++e2)
        o_s[w][(e2 + 8 * hi) * CC + h * 8 + an] = (_Float16)ov[e2];
    }
  }
  lds_fence();

  // ---- o @ Wo^T + bo, ReLU -> voxel-major f16 ----
  v8f acc2[4];
#pragma unroll
  for (int nt = 0; nt < 4; ++nt) acc2[nt] = zero8f();
#pragma unroll
  for (int chunk = 0; chunk < 2; ++chunk) {
    const _Float16* ap = &o_s[w][an * CC + chunk * 32 + hi * 16];
    v16h Ao = *(const v16h*)ap;
#pragma unroll
    for (int nt = 0; nt < 4; ++nt) {
      v16h Bm = *(const v16h*)&WoB[((chunk * 4 + nt) * 32 + lane) * 16];
      acc2[nt] = wmma16(Ao, Bm, acc2[nt]);
    }
  }
#pragma unroll
  for (int nt = 0; nt < 4; ++nt)
#pragma unroll
    for (int e2 = 0; e2 < 8; ++e2) {
      float vv = acc2[nt][e2] + bov[nt];
      q_s[w][(e2 + 8 * hi) * CC + nt * 16 + an] = (_Float16)(vv > 0.f ? vv : 0.f);
    }
  lds_fence();
  _Float16* og = ovm + ((size_t)(b * MVX + v0 + an)) * CC + hi * 32;
  *(v16h*)og        = *(const v16h*)&q_s[w][an * CC + hi * 32];
  *(v16h*)(og + 16) = *(const v16h*)&q_s[w][an * CC + hi * 32 + 16];
}

// ---------------------------------------------------------------------------
// 5) conv3d 3x3x3 circular, implicit GEMM (K = 27 taps x 64 ch = 1728),
//    + bc, ReLU, + x residual.  8 waves/block; wave = 16-voxel z-run x 64 co.
// ---------------------------------------------------------------------------
__global__ __launch_bounds__(256) void conv_kernel(
    const _Float16* __restrict__ ovm, const _Float16* __restrict__ WcB,
    const float* __restrict__ bc, const float* __restrict__ xin,
    float* __restrict__ out) {
  __shared__ float ot[8][16 * CC];     // 32 KB
  int w = threadIdx.x >> 5, lane = threadIdx.x & 31;
  int an = lane & 15, hi = lane >> 4;
  int T = blockIdx.x * 8 + w;
  int b = T >> 11;
  int r = T & 2047;
  int xx = r >> 6, yy = (r >> 1) & 31, z0 = (r & 1) * 16;

  v8f acc[4];
#pragma unroll
  for (int nt = 0; nt < 4; ++nt) acc[nt] = zero8f();

#pragma unroll 1
  for (int tap = 0; tap < 27; ++tap) {
    int dx = tap / 9 - 1, dy = (tap / 3) % 3 - 1, dz = tap % 3 - 1;
    int sx = (xx + dx) & 31, sy = (yy + dy) & 31, sz = (z0 + an + dz) & 31;
    const _Float16* ap0 =
        ovm + ((size_t)b * MVX + ((sx * 32 + sy) * 32 + sz)) * CC + hi * 16;
#pragma unroll
    for (int chunk = 0; chunk < 2; ++chunk) {
      const _Float16* ap = ap0 + chunk * 32;
      v16h A = *(const v16h*)ap;
#pragma unroll
      for (int nt = 0; nt < 4; ++nt) {
        v16h Bm = *(const v16h*)&WcB[(((tap * 2 + chunk) * 4 + nt) * 32 + lane) * 16];
        acc[nt] = wmma16(A, Bm, acc[nt]);
      }
    }
  }
#pragma unroll
  for (int nt = 0; nt < 4; ++nt) {
    float bcv = bc[nt * 16 + an];
#pragma unroll
    for (int e2 = 0; e2 < 8; ++e2) {
      float vv = acc[nt][e2] + bcv;
      ot[w][(e2 + 8 * hi) * CC + nt * 16 + an] = vv > 0.f ? vv : 0.f;
    }
  }
  lds_fence();
  int vbase = (xx * 32 + yy) * 32 + z0;
#pragma unroll
  for (int s = 0; s < 2; ++s) {
    int co = lane + 32 * s;
    const float* xp = xin + ((size_t)(b * CC + co)) * MVX + vbase;
    float* op = out + ((size_t)(b * CC + co)) * MVX + vbase;
#pragma unroll
    for (int g = 0; g < 4; ++g) {
      float4 xv = *(const float4*)(xp + g * 4);
      float4 rv;
      rv.x = ot[w][(g * 4 + 0) * CC + co] + xv.x;
      rv.y = ot[w][(g * 4 + 1) * CC + co] + xv.y;
      rv.z = ot[w][(g * 4 + 2) * CC + co] + xv.z;
      rv.w = ot[w][(g * 4 + 3) * CC + co] + xv.w;
      *(float4*)(op + g * 4) = rv;
    }
  }
}

// ---------------------------------------------------------------------------
extern "C" void kernel_launch(void* const* d_in, const int* in_sizes, int n_in,
                              void* d_out, int out_size, void* d_ws, size_t ws_size,
                              hipStream_t stream) {
  (void)in_sizes; (void)n_in; (void)out_size; (void)ws_size;
  const float* x   = (const float*)d_in[0];
  const float* mol = (const float*)d_in[1];
  const int*   bn  = (const int*)d_in[2];
  const float* Wq  = (const float*)d_in[3];
  const float* bq  = (const float*)d_in[4];
  const float* Wk  = (const float*)d_in[5];
  const float* bk  = (const float*)d_in[6];
  const float* Wv  = (const float*)d_in[7];
  const float* bv  = (const float*)d_in[8];
  const float* Wo  = (const float*)d_in[9];
  const float* bo  = (const float*)d_in[10];
  const float* Wc  = (const float*)d_in[11];
  const float* bc  = (const float*)d_in[12];
  float* out = (float*)d_out;

  _Float16* ws  = (_Float16*)d_ws;
  _Float16* xt  = ws;                                   // B*MV*64 halves
  _Float16* ovm = xt  + (size_t)BB * MVX * CC;          // B*MV*64
  _Float16* kf  = ovm + (size_t)BB * MVX * CC;          // B*64*64
  _Float16* vBp = kf  + (size_t)BB * NATOM * CC;        // B*8192
  _Float16* WqB = vBp + (size_t)BB * 8192;              // 4096
  _Float16* WoB = WqB + 4096;                           // 4096
  _Float16* WcB = WoB + 4096;                           // 110592

  prep_x<<<BB * MVX / 256, 256, 0, stream>>>(x, xt);
  prep_w<<<29, 256, 0, stream>>>(Wq, Wo, Wc, WqB, WoB, WcB);
  kv_kernel<<<1, 256, 0, stream>>>(mol, Wk, bk, Wv, bv, kf, vBp);
  attn_kernel<<<dim3(MVX / 64, BB), 128, 0, stream>>>(xt, kf, vBp, WqB, WoB,
                                                      bq, bo, bn, ovm);
  conv_kernel<<<BB * MVX / 16 / 8, 256, 0, stream>>>(ovm, WcB, bc, x, out);
}
